// SimpleHeteroGNN_9569187135976
// MI455X (gfx1250) — matrix operand order repaired and checked by reading, not compiled, not run on hardware
//
#include <hip/hip_runtime.h>
#include <hip/hip_bf16.h>

// Problem constants (match reference)
#define N_NODES 50000
#define N_EDGES 600000
#define DIM     128
#define GRAPHS  512
#define LAYERS  3
#define OUTC    16
#define BN_EPS  1e-5f

typedef __attribute__((ext_vector_type(16))) __bf16 v16bf;
typedef __attribute__((ext_vector_type(8)))  __bf16 v8bf;
typedef __attribute__((ext_vector_type(8)))  float  v8f;

// ---------------- elementwise / utility kernels ----------------

__global__ void copy_kernel(float* __restrict__ dst, const float* __restrict__ src, long long n) {
  long long i = (long long)blockIdx.x * blockDim.x + threadIdx.x;
  if (i < n) dst[i] = src[i];
}

__global__ void zero_kernel(float* __restrict__ p, long long n) {
  long long i = (long long)blockIdx.x * blockDim.x + threadIdx.x;
  if (i < n) p[i] = 0.f;
}

// agg[dst[e]][f] += h[src[e]][f]  (agg pre-initialized to h: GIN eps=0 self term)
__global__ void scatter_kernel(const float* __restrict__ h, const int* __restrict__ src,
                               const int* __restrict__ dst, float* __restrict__ agg) {
  long long i = (long long)blockIdx.x * blockDim.x + threadIdx.x;
  long long total = (long long)N_EDGES * DIM;
  if (i >= total) return;
  int e = (int)(i >> 7);
  int f = (int)(i & 127);
  atomicAdd(&agg[(size_t)dst[e] * DIM + f], h[(size_t)src[e] * DIM + f]);
}

// per-column sum and sum-of-squares over nrows (DIM=128 columns); sums[0..127]=sum, sums[128..255]=sumsq
__global__ void stats_kernel(const float* __restrict__ Z, float* __restrict__ sums, int nrows) {
  int f = threadIdx.x & 127;
  int r = blockIdx.x * 2 + (threadIdx.x >> 7);
  float s = 0.f, s2 = 0.f;
  int stride = gridDim.x * 2;
  for (; r < nrows; r += stride) {
    float v = Z[(size_t)r * DIM + f];
    s += v;
    s2 += v * v;
  }
  atomicAdd(&sums[f], s);
  atomicAdd(&sums[DIM + f], s2);
}

// Z = relu(gamma*(Z-mean)*rsqrt(var+eps)+beta), batch stats from sums buffer
__global__ void bnrelu_kernel(float* __restrict__ Z, const float* __restrict__ sums,
                              const float* __restrict__ gamma, const float* __restrict__ beta,
                              int nrows) {
  long long i = (long long)blockIdx.x * blockDim.x + threadIdx.x;
  long long total = (long long)nrows * DIM;
  if (i >= total) return;
  int f = (int)(i & 127);
  float inv = 1.0f / (float)nrows;
  float m = sums[f] * inv;
  float var = sums[DIM + f] * inv - m * m;
  float o = gamma[f] * (Z[i] - m) * rsqrtf(var + BN_EPS) + beta[f];
  Z[i] = o > 0.f ? o : 0.f;
}

__global__ void pool_accum_kernel(const float* __restrict__ h, const int* __restrict__ batch,
                                  float* __restrict__ pooled, float* __restrict__ counts) {
  long long i = (long long)blockIdx.x * blockDim.x + threadIdx.x;
  long long total = (long long)N_NODES * DIM;
  if (i >= total) return;
  int node = (int)(i >> 7);
  int f = (int)(i & 127);
  int g = batch[node];
  atomicAdd(&pooled[(size_t)g * DIM + f], h[i]);
  if (f == 0) atomicAdd(&counts[g], 1.0f);
}

__global__ void pool_div_kernel(float* __restrict__ pooled, const float* __restrict__ counts) {
  int i = blockIdx.x * blockDim.x + threadIdx.x;
  if (i >= GRAPHS * DIM) return;
  float c = counts[i >> 7];
  pooled[i] /= fmaxf(c, 1.0f);
}

// ---------------- WMMA GEMM:  Z[nrows x (NTILES*16)] = A[nrows x 128] @ W[128 x (NTILES*16)] + bias
// Block = 256 threads (8 waves); each wave owns a 16-row strip, block covers 128 rows.
// W is staged transposed in LDS as bf16 (stride 136 to avoid bank conflicts).
// All A loads for the full K=128 row are issued up front into distinct registers
// (16 b128 loads in flight), then converted fp32->bf16 into 4 read-only fragments,
// so the MMA phase has no WAR hazards on A and no per-chunk load stalls.
template <int NTILES>
__global__ void gemm_bias_wmma(const float* __restrict__ A, const float* __restrict__ W,
                               const float* __restrict__ bias, float* __restrict__ Z,
                               int nrows) {
  const int ncols = NTILES * 16;
  __shared__ __bf16 sW[128 * 136];

  int tid = threadIdx.x;
  // stage W^T into LDS as bf16: sW[c*136 + k] = W[k*ncols + c]
  for (int idx = tid; idx < 128 * ncols; idx += 256) {
    int k = idx / ncols;
    int c = idx - k * ncols;
    sW[c * 136 + k] = (__bf16)W[idx];
  }
  __syncthreads();

  int wid = tid >> 5;
  int lane = tid & 31;
  int half = lane >> 4;   // 0: lanes 0-15, 1: lanes 16-31
  int m16 = lane & 15;
  int rowBase = blockIdx.x * 128 + wid * 16;
  if (rowBase >= nrows) return;  // wave-uniform exit; EXEC stays all-ones for WMMA

  int row = rowBase + m16;
  if (row > nrows - 1) row = nrows - 1;  // clamp (stores are guarded)
  const float* arow = A + (size_t)row * DIM;

  // ---- issue ALL A loads up front (16 independent b128 loads) ----
  // A fragment (16-bit A 16x32 layout): elems 0-7  -> K = kc*32 + 8*half + {0..7}
  //                                     elems 8-15 -> K = kc*32 + 16 + 8*half + {0..7}
  float4 af[4][4];
#pragma unroll
  for (int kc = 0; kc < 4; kc++) {
    int kb = kc * 32 + half * 8;
    af[kc][0] = *(const float4*)(arow + kb);
    af[kc][1] = *(const float4*)(arow + kb + 4);
    af[kc][2] = *(const float4*)(arow + kb + 16);
    af[kc][3] = *(const float4*)(arow + kb + 20);
  }
  // ---- convert to 4 bf16 fragments (read-only for the MMA phase) ----
  v16bf afr[4];
#pragma unroll
  for (int kc = 0; kc < 4; kc++) {
#pragma unroll
    for (int q = 0; q < 4; q++) {
      afr[kc][4 * q + 0] = (__bf16)af[kc][q].x;
      afr[kc][4 * q + 1] = (__bf16)af[kc][q].y;
      afr[kc][4 * q + 2] = (__bf16)af[kc][q].z;
      afr[kc][4 * q + 3] = (__bf16)af[kc][q].w;
    }
  }

  v8f acc[NTILES];
#pragma unroll
  for (int nt = 0; nt < NTILES; nt++) {
#pragma unroll
    for (int r = 0; r < 8; r++) acc[nt][r] = 0.f;
  }

#pragma unroll
  for (int kc = 0; kc < 4; kc++) {
#pragma unroll
    for (int nt = 0; nt < NTILES; nt++) {
      // B fragment (32x16): lane holds column c = nt*16+m16, elems 0-7 K = kc*32 + 8*half + {0..7}, elems 8-15 K = kc*32 + 16 + 8*half + {0..7}
      const __bf16* bp = &sW[(nt * 16 + m16) * 136 + kc * 32 + half * 8];
      v8bf blo = *(const v8bf*)bp;
      v8bf bhi = *(const v8bf*)(bp + 16);
      v16bf b;
#pragma unroll
      for (int e = 0; e < 8; e++) { b[e] = blo[e]; b[e + 8] = bhi[e]; }

      acc[nt] = __builtin_amdgcn_wmma_f32_16x16x32_bf16(
          /*neg_a=*/false, afr[kc], /*neg_b=*/false, b,
          /*c_mod=*/(short)0, acc[nt], /*reuse_a=*/false, /*reuse_b=*/false);
    }
  }

  // C/D layout: VGPR r -> row M = r + 8*half; lane -> col = nt*16 + m16
#pragma unroll
  for (int nt = 0; nt < NTILES; nt++) {
    int c = nt * 16 + m16;
    float bs = bias[c];
#pragma unroll
    for (int r = 0; r < 8; r++) {
      int rr = rowBase + half * 8 + r;
      if (rr < nrows) Z[(size_t)rr * ncols + c] = acc[nt][r] + bs;
    }
  }
}

// ---------------- host launch ----------------

static inline int cdiv_ll(long long a, long long b) { return (int)((a + b - 1) / b); }

extern "C" void kernel_launch(void* const* d_in, const int* in_sizes, int n_in,
                              void* d_out, int out_size, void* d_ws, size_t ws_size,
                              hipStream_t stream) {
  const float* x    = (const float*)d_in[0];
  const int*   ei   = (const int*)d_in[1];
  const int*   srcI = ei;
  const int*   dstI = ei + N_EDGES;
  const int*   batch = (const int*)d_in[2];
  const float* W1s  = (const float*)d_in[3];
  const float* b1s  = (const float*)d_in[4];
  const float* g1s  = (const float*)d_in[5];
  const float* be1s = (const float*)d_in[6];
  const float* W2s  = (const float*)d_in[7];
  const float* b2s  = (const float*)d_in[8];
  const float* gns  = (const float*)d_in[9];
  const float* bns  = (const float*)d_in[10];
  const float* Wc1  = (const float*)d_in[11];
  const float* bc1  = (const float*)d_in[12];
  const float* gc   = (const float*)d_in[13];
  const float* bec  = (const float*)d_in[14];
  const float* Wc2  = (const float*)d_in[15];
  const float* bc2  = (const float*)d_in[16];
  float* out = (float*)d_out;

  // workspace layout (floats): agg | t1 | t2 | pooled | pc1 | counts | sums  (~77.5 MB)
  float* ws = (float*)d_ws;
  size_t NH = (size_t)N_NODES * DIM;
  float* agg    = ws;
  float* t1     = agg + NH;
  float* t2     = t1 + NH;
  float* pooled = t2 + NH;
  float* pc1    = pooled + (size_t)GRAPHS * DIM;
  float* counts = pc1 + (size_t)GRAPHS * DIM;
  float* sums   = counts + GRAPHS;

  long long nh = (long long)NH;
  int gridElem = cdiv_ll(nh, 256);
  int gridScat = cdiv_ll((long long)N_EDGES * DIM, 256);
  int gridGemmN = cdiv_ll(N_NODES, 128);
  int gridGemmG = cdiv_ll(GRAPHS, 128);

  const float* h = x;
  for (int l = 0; l < LAYERS; l++) {
    // agg = h + sum_{j->i} h_j
    copy_kernel<<<gridElem, 256, 0, stream>>>(agg, h, nh);
    scatter_kernel<<<gridScat, 256, 0, stream>>>(h, srcI, dstI, agg);
    // z = agg @ W1 + b1 ; BN ; ReLU
    gemm_bias_wmma<8><<<gridGemmN, 256, 0, stream>>>(agg, W1s + (size_t)l * DIM * DIM,
                                                     b1s + l * DIM, t1, N_NODES);
    zero_kernel<<<1, 256, 0, stream>>>(sums, 256);
    stats_kernel<<<256, 256, 0, stream>>>(t1, sums, N_NODES);
    bnrelu_kernel<<<gridElem, 256, 0, stream>>>(t1, sums, g1s + l * DIM, be1s + l * DIM, N_NODES);
    // z = z @ W2 + b2 ; BN ; ReLU  -> next h
    gemm_bias_wmma<8><<<gridGemmN, 256, 0, stream>>>(t1, W2s + (size_t)l * DIM * DIM,
                                                     b2s + l * DIM, t2, N_NODES);
    zero_kernel<<<1, 256, 0, stream>>>(sums, 256);
    stats_kernel<<<256, 256, 0, stream>>>(t2, sums, N_NODES);
    bnrelu_kernel<<<gridElem, 256, 0, stream>>>(t2, sums, gns + l * DIM, bns + l * DIM, N_NODES);
    h = t2;
  }

  // global mean pool per graph
  zero_kernel<<<cdiv_ll((long long)GRAPHS * DIM, 256), 256, 0, stream>>>(pooled, (long long)GRAPHS * DIM);
  zero_kernel<<<cdiv_ll(GRAPHS, 256), 256, 0, stream>>>(counts, GRAPHS);
  pool_accum_kernel<<<gridElem, 256, 0, stream>>>(h, batch, pooled, counts);
  pool_div_kernel<<<cdiv_ll((long long)GRAPHS * DIM, 256), 256, 0, stream>>>(pooled, counts);

  // classifier: pooled @ Wc1 + bc1 ; BN ; ReLU ; @ Wc2 + bc2 -> out [512 x 16]
  gemm_bias_wmma<8><<<gridGemmG, 256, 0, stream>>>(pooled, Wc1, bc1, pc1, GRAPHS);
  zero_kernel<<<1, 256, 0, stream>>>(sums, 256);
  stats_kernel<<<256, 256, 0, stream>>>(pc1, sums, GRAPHS);
  bnrelu_kernel<<<cdiv_ll((long long)GRAPHS * DIM, 256), 256, 0, stream>>>(pc1, sums, gc, bec, GRAPHS);
  gemm_bias_wmma<1><<<gridGemmG, 256, 0, stream>>>(pc1, Wc2, bc2, out, GRAPHS);
}
